// YatNMN_30365418783503
// MI455X (gfx1250) — matrix operand order repaired and checked by reading, not compiled, not run on hardware
//
#include <hip/hip_runtime.h>

// ---------------------------------------------------------------------------
// YatNMN: out = scale * ( (x@W)^2 / (||x||^2 + ||w||^2 - 2 x@W + eps) + bias )
// M=16384, N=8192, K=2048, all f32.
// Compute-bound (AI ~745 flop/byte) -> use bf16 WMMA with 3-term hi/lo split
// for ~fp32 accuracy at ~3x the f32-WMMA instruction efficiency... see notes.
// ---------------------------------------------------------------------------

typedef __bf16 bf16_t;
typedef bf16_t v16bf __attribute__((ext_vector_type(16)));
typedef bf16_t v8bf  __attribute__((ext_vector_type(8)));
typedef float  v8f   __attribute__((ext_vector_type(8)));
typedef unsigned int u32;
typedef u32 v4u __attribute__((ext_vector_type(4)));
typedef u32 v8u __attribute__((ext_vector_type(8)));

#define M_DIM 16384
#define N_DIM 8192
#define K_DIM 2048
#define BM 128
#define BN 128
#define BK 32
#define A_STR 40   // bf16 elems per LDS A row (32 used + 8 pad) -> conflict-free b128 reads
#define B_STR 36   // dwords per LDS B row (32 used + 4 pad)     -> conflict-free b128 reads
#define EPSV (1.0f / 137.0f)

__device__ __forceinline__ u32 f2bf(float f) {
  u32 u = __builtin_bit_cast(u32, f);
  return (u + 0x7FFFu + ((u >> 16) & 1u)) >> 16;   // RNE f32->bf16 (finite inputs)
}
__device__ __forceinline__ float bf2f(u32 b) {
  return __builtin_bit_cast(float, b << 16);
}

// ------------------------- rowsq: ||x_m||^2 (exact f32) ---------------------
__global__ __launch_bounds__(256)
void yat_rowsq_kernel(const float* __restrict__ A, float* __restrict__ rowsq) {
  int row  = blockIdx.x * 8 + (threadIdx.x >> 5);   // one wave per row
  int lane = threadIdx.x & 31;
  const float4* p = (const float4*)(A + (size_t)row * K_DIM);
  float s = 0.0f;
#pragma unroll 4
  for (int i = lane; i < K_DIM / 4; i += 32) {
    float4 v = p[i];
    s += v.x * v.x + v.y * v.y + v.z * v.z + v.w * v.w;
  }
#pragma unroll
  for (int off = 16; off > 0; off >>= 1) s += __shfl_xor(s, off, 32);
  if (lane == 0) rowsq[row] = s;
}

// ------------------------- colsq: ||w_n||^2 (exact f32) ---------------------
__global__ __launch_bounds__(256)
void yat_colsq_kernel(const float* __restrict__ B, float* __restrict__ colsq) {
  int n = blockIdx.x * blockDim.x + threadIdx.x;
  const float* p = B + n;
  float s = 0.0f;
#pragma unroll 8
  for (int k = 0; k < K_DIM; ++k) {
    float v = p[(size_t)k * N_DIM];
    s += v * v;
  }
  colsq[n] = s;
}

// ------------------------------- main GEMM ----------------------------------
__global__ __launch_bounds__(256)
void yat_gemm_kernel(const float* __restrict__ A, const float* __restrict__ B,
                     const float* __restrict__ bias, const float* __restrict__ alpha,
                     const float* __restrict__ rowsq, const float* __restrict__ colsq,
                     float* __restrict__ out)
{
  __shared__ bf16_t sAh[BM * A_STR];
  __shared__ bf16_t sAl[BM * A_STR];
  __shared__ u32    sB [BN * B_STR];    // packed (hi16<<16)|lo16, layout [n][k]
  __shared__ float  sRow[BM];
  __shared__ float  sCol[BN];
  __shared__ float  sBias[BN];

  const int tid   = threadIdx.x;
  const int lane  = tid & 31;
  const int wave  = tid >> 5;
  const int lmod  = lane & 15;
  const int lhalf = lane >> 4;
  const int gm0   = blockIdx.y * BM;
  const int gn0   = blockIdx.x * BN;
  const int wm    = (wave & 1) * 64;     // wave tile: 64(M) x 32(N)
  const int wn    = (wave >> 1) * 32;

  if (tid < 128) {
    sRow[tid] = rowsq[gm0 + tid];
  } else {
    int c = tid - 128;
    sCol[c]  = colsq[gn0 + c];
    sBias[c] = bias[gn0 + c];
  }

  // per-thread global staging assignment
  const int aRow  = tid >> 1;            // 0..127
  const int aK    = (tid & 1) * 16;      // 0 or 16
  const int bkRow = tid >> 3;            // 0..31
  const int bnCol = (tid & 7) * 16;      // 0..112

  const float* pAg = A + (size_t)(gm0 + aRow) * K_DIM + aK;
  const float* pBg = B + (size_t)bkRow * N_DIM + (gn0 + bnCol);

  float4 pa[4], pb[4];
#pragma unroll
  for (int i = 0; i < 4; ++i) pa[i] = ((const float4*)pAg)[i];
#pragma unroll
  for (int i = 0; i < 4; ++i) pb[i] = ((const float4*)pBg)[i];

  v8f acc[4][2];
#pragma unroll
  for (int mi = 0; mi < 4; ++mi)
#pragma unroll
    for (int ni = 0; ni < 2; ++ni)
#pragma unroll
      for (int r = 0; r < 8; ++r) acc[mi][ni][r] = 0.0f;

  for (int kb = 0; kb < K_DIM; kb += BK) {
    // ---- stage current tile into LDS (f32 -> bf16 hi/lo) ----
    {
      float fa[16];
#pragma unroll
      for (int i = 0; i < 4; ++i) {
        fa[4*i+0] = pa[i].x; fa[4*i+1] = pa[i].y; fa[4*i+2] = pa[i].z; fa[4*i+3] = pa[i].w;
      }
      u32 hw[8], lw[8];
#pragma unroll
      for (int j = 0; j < 8; ++j) {
        u32 h0 = f2bf(fa[2*j]);     u32 l0 = f2bf(fa[2*j]   - bf2f(h0));
        u32 h1 = f2bf(fa[2*j+1]);   u32 l1 = f2bf(fa[2*j+1] - bf2f(h1));
        hw[j] = h0 | (h1 << 16);
        lw[j] = l0 | (l1 << 16);
      }
      bf16_t* dah = &sAh[aRow * A_STR + aK];
      bf16_t* dal = &sAl[aRow * A_STR + aK];
      *(v4u*)(dah)     = (v4u){hw[0], hw[1], hw[2], hw[3]};
      *(v4u*)(dah + 8) = (v4u){hw[4], hw[5], hw[6], hw[7]};
      *(v4u*)(dal)     = (v4u){lw[0], lw[1], lw[2], lw[3]};
      *(v4u*)(dal + 8) = (v4u){lw[4], lw[5], lw[6], lw[7]};

      float fb[16];
#pragma unroll
      for (int i = 0; i < 4; ++i) {
        fb[4*i+0] = pb[i].x; fb[4*i+1] = pb[i].y; fb[4*i+2] = pb[i].z; fb[4*i+3] = pb[i].w;
      }
#pragma unroll
      for (int jj = 0; jj < 16; ++jj) {
        int j = (jj + tid) & 15;                 // rotate -> conflict-free scatter
        u32 h = f2bf(fb[j]);
        u32 l = f2bf(fb[j] - bf2f(h));
        sB[(bnCol + j) * B_STR + bkRow] = (h << 16) | l;
      }
    }
    __syncthreads();

    // ---- software pipeline: issue next tile's global loads now ----
    if (kb + BK < K_DIM) {
      const float4* na = (const float4*)(pAg + (kb + BK));
      const float4* nb = (const float4*)(pBg + (size_t)(kb + BK) * N_DIM);
#pragma unroll
      for (int i = 0; i < 4; ++i) pa[i] = na[i];
#pragma unroll
      for (int i = 0; i < 4; ++i) pb[i] = nb[i];
    }

    // ---- compute: 24 WMMAs / wave / iter ----
    v16bf bh[2], bl[2];
#pragma unroll
    for (int ni = 0; ni < 2; ++ni) {
      const u32* p = &sB[(wn + ni * 16 + lmod) * B_STR + 16 * lhalf];
      v4u q0 = *(const v4u*)(p + 0);
      v4u q1 = *(const v4u*)(p + 4);
      v4u q2 = *(const v4u*)(p + 8);
      v4u q3 = *(const v4u*)(p + 12);
      u32 d[16] = {q0[0],q0[1],q0[2],q0[3], q1[0],q1[1],q1[2],q1[3],
                   q2[0],q2[1],q2[2],q2[3], q3[0],q3[1],q3[2],q3[3]};
      v8u hh, ll;
#pragma unroll
      for (int j = 0; j < 8; ++j) {
        hh[j] = __builtin_amdgcn_perm(d[2*j+1], d[2*j], 0x07060302u);  // hi halves
        ll[j] = __builtin_amdgcn_perm(d[2*j+1], d[2*j], 0x05040100u);  // lo halves
      }
      bh[ni] = __builtin_bit_cast(v16bf, hh);
      bl[ni] = __builtin_bit_cast(v16bf, ll);
    }

#pragma unroll
    for (int mi = 0; mi < 4; ++mi) {
      const bf16_t* pah = &sAh[(wm + mi * 16 + lmod) * A_STR + 16 * lhalf];
      const bf16_t* pal = &sAl[(wm + mi * 16 + lmod) * A_STR + 16 * lhalf];
      v8bf ah0 = *(const v8bf*)pah;
      v8bf ah1 = *(const v8bf*)(pah + 8);
      v8bf al0 = *(const v8bf*)pal;
      v8bf al1 = *(const v8bf*)(pal + 8);
      v16bf ah, al;
#pragma unroll
      for (int i = 0; i < 8; ++i) {
        ah[i] = ah0[i]; ah[8 + i] = ah1[i];
        al[i] = al0[i]; al[8 + i] = al1[i];
      }
#pragma unroll
      for (int ni = 0; ni < 2; ++ni) {
        acc[mi][ni] = __builtin_amdgcn_wmma_f32_16x16x32_bf16(
            false, ah, false, bh[ni], (short)0, acc[mi][ni], false, false);
        acc[mi][ni] = __builtin_amdgcn_wmma_f32_16x16x32_bf16(
            false, ah, false, bl[ni], (short)0, acc[mi][ni], false, false);
        acc[mi][ni] = __builtin_amdgcn_wmma_f32_16x16x32_bf16(
            false, al, false, bh[ni], (short)0, acc[mi][ni], false, false);
      }
    }
    __syncthreads();
  }

  // ---- epilogue: y^2 / (||x||^2 + ||w||^2 - 2y + eps), bias, scale ----
  const float sc = __powf(90.50966799f / log1pf(8192.0f), alpha[0]);
#pragma unroll
  for (int mi = 0; mi < 4; ++mi) {
#pragma unroll
    for (int ni = 0; ni < 2; ++ni) {
      const int nl = wn + ni * 16 + lmod;
      const int gn = gn0 + nl;
      const float cs = sCol[nl];
      const float bb = sBias[nl];
#pragma unroll
      for (int r = 0; r < 8; ++r) {
        const int ml = wm + mi * 16 + r + 8 * lhalf;   // C/D layout: VGPR r -> M=r+8*half
        const float y = acc[mi][ni][r];
        const float rs = sRow[ml];
        const float denom = rs + cs - 2.0f * y + EPSV;
        const float o = (y * y / denom + bb) * sc;
        out[(size_t)(gm0 + ml) * N_DIM + gn] = o;
      }
    }
  }
}

// ---------------------------------------------------------------------------
extern "C" void kernel_launch(void* const* d_in, const int* in_sizes, int n_in,
                              void* d_out, int out_size, void* d_ws, size_t ws_size,
                              hipStream_t stream) {
  const float* x     = (const float*)d_in[0];   // [8,2048,2048] -> [16384,2048]
  const float* w     = (const float*)d_in[1];   // [2048,8192]
  const float* bias  = (const float*)d_in[2];   // [8192]
  const float* alpha = (const float*)d_in[3];   // [1]
  float* out = (float*)d_out;

  float* rowsq = (float*)d_ws;          // 16384 floats
  float* colsq = rowsq + M_DIM;         // 8192 floats (96 KB total scratch)

  yat_rowsq_kernel<<<M_DIM / 8, 256, 0, stream>>>(x, rowsq);
  yat_colsq_kernel<<<N_DIM / 256, 256, 0, stream>>>(w, colsq);

  dim3 grid(N_DIM / BN, M_DIM / BM);    // (64, 128)
  yat_gemm_kernel<<<grid, 256, 0, stream>>>(x, w, bias, alpha, rowsq, colsq, out);
}
